// AttentionModel_6837587936208
// MI455X (gfx1250) — matrix operand order repaired and checked by reading, not compiled, not run on hardware
//
#include <hip/hip_runtime.h>
#include <hip/hip_bf16.h>
#include <math.h>

typedef __attribute__((ext_vector_type(16))) _Float16 v16h;
typedef __attribute__((ext_vector_type(8)))  _Float16 v8h;
typedef __attribute__((ext_vector_type(8)))  float    v8f;

#define DEV_INLINE __device__ __forceinline__

constexpr int Bc  = 512;   // batch
constexpr int Lc  = 60;    // seq len
constexpr int Dc  = 300;   // embed dim
constexpr int DPc = 320;   // embed dim padded to multiple of 32
constexpr int Hc  = 256;   // hidden
constexpr int Gc  = 1024;  // 4*H gates
constexpr int KT  = DPc + Hc;  // fused LSTM K = 576

DEV_INLINE float sigm(float x) { return 1.0f / (1.0f + __expf(-x)); }

// ---- WMMA fragment helpers (ISA 7.12.2 layouts, wave32) -------------------
// A: 16x32 (MxK) f16 row-major, stride lda (halves). Per lane the fragment is
// two contiguous 8-half runs: k = half*8+0..7 and k = 16+half*8+0..7.
DEV_INLINE v16h frag_a(const _Float16* __restrict__ A, int lda, int lane) {
  const int half = (lane >> 4) & 1, m = lane & 15;
  const _Float16* p = A + (size_t)m * lda + half * 8;
  v8h lo = *(const v8h*)(p);
  v8h hi = *(const v8h*)(p + 16);
  return __builtin_shufflevector(lo, hi, 0, 1, 2, 3, 4, 5, 6, 7,
                                 8, 9, 10, 11, 12, 13, 14, 15);
}

// B: 32x16 (KxN). Lane n holds K = half*8+0..7 and 16+half*8+0..7 of column n
// (same order as A) -> with the weight stored TRANSPOSED as BT[N][K] (K
// contiguous, stride ldk halves) the per-lane fragment is two 16-byte loads.
DEV_INLINE v16h frag_bT(const _Float16* __restrict__ BT, int ldk, int lane) {
  const int half = (lane >> 4) & 1, n = lane & 15;
  const _Float16* p = BT + (size_t)n * ldk + half * 8;
  v8h lo = *(const v8h*)(p);
  v8h hi = *(const v8h*)(p + 16);
  return __builtin_shufflevector(lo, hi, 0, 1, 2, 3, 4, 5, 6, 7,
                                 8, 9, 10, 11, 12, 13, 14, 15);
}

DEV_INLINE v8f wmma_f16(v16h a, v16h b, v8f c) {
  return __builtin_amdgcn_wmma_f32_16x16x32_f16(false, a, false, b, (short)0, c,
                                                false, false);
}

// ---- weight preparation kernels -------------------------------------------

// Fused transposed LSTM weight: WT[n][k], n<1024, k<576.
// k<300 -> Wx[k][n]; 300<=k<320 -> 0 (pad); k>=320 -> Wh[k-320][n].
__global__ void build_wT_kernel(const float* __restrict__ Wx,
                                const float* __restrict__ Wh,
                                _Float16* __restrict__ WT) {
  int idx = blockIdx.x * blockDim.x + threadIdx.x;
  if (idx >= Gc * KT) return;
  const int n = idx / KT, k = idx % KT;
  float v = 0.0f;
  if (k < Dc)        v = Wx[(size_t)k * Gc + n];
  else if (k >= DPc) v = Wh[(size_t)(k - DPc) * Gc + n];
  WT[idx] = (_Float16)v;
}

// Transpose+convert an H x H f32 matrix: D[n][k] = S[k][n], f16.
__global__ void cvtT_kernel(const float* __restrict__ S, _Float16* __restrict__ D) {
  int idx = blockIdx.x * blockDim.x + threadIdx.x;
  if (idx >= Hc * Hc) return;
  const int n = idx >> 8, k = idx & 255;
  D[idx] = (_Float16)S[(size_t)k * Hc + n];
}

__global__ void embed_kernel(const float* __restrict__ E, const int* __restrict__ tok,
                             _Float16* __restrict__ X) {
  int idx = blockIdx.x * blockDim.x + threadIdx.x;
  if (idx >= Bc * Lc * DPc) return;
  const int d = idx % DPc;
  const int bl = idx / DPc;
  float v = 0.0f;
  if (d < Dc) v = E[(size_t)tok[bl] * Dc + d];
  X[idx] = (_Float16)v;
}

// ---- fused LSTM: one workgroup = 16 batch rows, all 60 timesteps ----------
// gates = [x_t | h] @ WT^T + bias, gate order i,j,f,o (forget_bias = 1).
// Wave w owns h-units [32w,32w+32): acc[q>>2][q&3] = gate block (q&3) for
// column-tile ct = 2w+(q>>2). K loop split x-part/h-part (no branchy select);
// all 8 B fragments loaded as one clause before the 8 back-to-back WMMAs.
__global__ __launch_bounds__(256) void lstm_kernel(
    const _Float16* __restrict__ X,   // [B, L, DP] f16
    const _Float16* __restrict__ WT,  // [G, KT]    f16 transposed fused weight
    const float* __restrict__ bias,   // [G]        f32
    const int* __restrict__ seqlen,   // [B]
    _Float16* __restrict__ Y,         // [B, L, H]  f16 (masked output)
    _Float16* __restrict__ hlast) {   // [B, H]     f16 (frozen final h)
  __shared__ alignas(32) _Float16 sX[16 * DPc];
  __shared__ alignas(32) _Float16 sH[16 * Hc];
  __shared__ float sC[16 * Hc];

  const int tid  = threadIdx.x;
  const int wave = tid >> 5, lane = tid & 31;
  const int half = (lane >> 4) & 1, nn = lane & 15;
  const int b0   = blockIdx.x * 16;

  for (int i = tid; i < 16 * Hc; i += 256) { sH[i] = (_Float16)0.0f; sC[i] = 0.0f; }
  __syncthreads();

  for (int t = 0; t < Lc; ++t) {
    // stage x_t tile [16 x DP] (vectorized: 8 halves per thread)
    for (int i = tid; i < 16 * DPc / 8; i += 256) {
      const int row = i / (DPc / 8), c8 = i % (DPc / 8);
      *(v8h*)(sX + row * DPc + c8 * 8) =
          *(const v8h*)(X + (size_t)(b0 + row) * Lc * DPc + (size_t)t * DPc + c8 * 8);
    }
    __syncthreads();

    v8f acc[8] = {};
    // x part: K = 0..319 from sX
#pragma unroll 2
    for (int ks = 0; ks < DPc / 32; ++ks) {
      const int k = ks * 32;
      v16h a = frag_a(sX + k, DPc, lane);
      v16h bf[8];
#pragma unroll
      for (int q = 0; q < 8; ++q) {
        const int ct = 2 * wave + (q >> 2), g = q & 3;
        bf[q] = frag_bT(WT + (size_t)(g * Hc + ct * 16) * KT + k, KT, lane);
      }
#pragma unroll
      for (int q = 0; q < 8; ++q) acc[q] = wmma_f16(a, bf[q], acc[q]);
    }
    // h part: K = 320..575 from sH
#pragma unroll 2
    for (int ks = 0; ks < Hc / 32; ++ks) {
      const int k = ks * 32;
      v16h a = frag_a(sH + k, Hc, lane);
      v16h bf[8];
#pragma unroll
      for (int q = 0; q < 8; ++q) {
        const int ct = 2 * wave + (q >> 2), g = q & 3;
        bf[q] = frag_bT(WT + (size_t)(g * Hc + ct * 16) * KT + DPc + k, KT, lane);
      }
#pragma unroll
      for (int q = 0; q < 8; ++q) acc[q] = wmma_f16(a, bf[q], acc[q]);
    }
    __syncthreads();  // all waves done reading sH/sX

    // gate nonlinearity + state update + masked output
#pragma unroll
    for (int j2 = 0; j2 < 2; ++j2) {
      const int u = (2 * wave + j2) * 16 + nn;
#pragma unroll
      for (int v = 0; v < 8; ++v) {
        const int m = v + 8 * half;
        const int b = b0 + m;
        const float gi = acc[4 * j2 + 0][v] + bias[0 * Hc + u];
        const float gj = acc[4 * j2 + 1][v] + bias[1 * Hc + u];
        const float gf = acc[4 * j2 + 2][v] + bias[2 * Hc + u];
        const float go = acc[4 * j2 + 3][v] + bias[3 * Hc + u];
        const float c_old = sC[m * Hc + u];
        const float c_new = c_old * sigm(gf + 1.0f) + sigm(gi) * tanhf(gj);
        const float h_new = tanhf(c_new) * sigm(go);
        _Float16* yp = Y + (size_t)b * Lc * Hc + (size_t)t * Hc + u;
        if (t < seqlen[b]) {
          sC[m * Hc + u] = c_new;
          sH[m * Hc + u] = (_Float16)h_new;
          *yp = (_Float16)h_new;
        } else {
          *yp = (_Float16)0.0f;
        }
      }
    }
    __syncthreads();
  }

  for (int i = tid; i < 16 * Hc; i += 256) {
    const int row = i / Hc, u = i % Hc;
    hlast[(size_t)(b0 + row) * Hc + u] = sH[i];
  }
}

// ---- generic wave-per-tile WMMA GEMM: C[f32] = A[f16] x BT[f16]^T ---------
// Two interleaved accumulator chains (even/odd k-steps) for WMMA ILP.
__global__ __launch_bounds__(32) void gemm_f16f32_kernel(
    const _Float16* __restrict__ A, const _Float16* __restrict__ BT,
    float* __restrict__ C, int K, int lda, int ldk, int ldc) {
  const int lane = threadIdx.x & 31;
  const int nt = blockIdx.x, mt = blockIdx.y;
  v8f acc0 = {}, acc1 = {};
  const _Float16* Ab = A + (size_t)mt * 16 * lda;
  const _Float16* Bb = BT + (size_t)nt * 16 * ldk;
  for (int k = 0; k < K; k += 64) {
    v16h a0 = frag_a(Ab + k, lda, lane);
    v16h b0 = frag_bT(Bb + k, ldk, lane);
    v16h a1 = frag_a(Ab + k + 32, lda, lane);
    v16h b1 = frag_bT(Bb + k + 32, ldk, lane);
    acc0 = wmma_f16(a0, b0, acc0);
    acc1 = wmma_f16(a1, b1, acc1);
  }
  const int half = (lane >> 4) & 1, nn = lane & 15;
#pragma unroll
  for (int v = 0; v < 8; ++v)
    C[(size_t)(mt * 16 + v + 8 * half) * ldc + nt * 16 + nn] = acc0[v] + acc1[v];
}

// ---- fused word-by-word attention scan ------------------------------------
__global__ __launch_bounds__(256) void attend_kernel(
    const _Float16* __restrict__ Y,    // [B,L,H] f16 (scored / weighted seq)
    const _Float16* __restrict__ Y2,   // [B,L,H] f16 (h_t source)
    const float* __restrict__ WyY,     // [B,L,H] f32
    const _Float16* __restrict__ WhaT, // [H,H] f16, transposed
    const _Float16* __restrict__ WraT, const _Float16* __restrict__ WtaT,
    const float* __restrict__ wa,      // [H]
    const int* __restrict__ s_mask,    // seqlen for softmax mask
    const int* __restrict__ s_sel,     // seqlen selecting r_L
    _Float16* __restrict__ rL) {       // [B,H] f16
  __shared__ alignas(32) _Float16 sRh[16 * Hc];
  __shared__ float sTmp[16 * Hc];
  __shared__ float sTT[16 * Hc];
  __shared__ float sScore[16 * 64];

  const int tid  = threadIdx.x;
  const int wave = tid >> 5, lane = tid & 31;
  const int half = (lane >> 4) & 1, nn = lane & 15;
  const int b0   = blockIdx.x * 16;

  for (int i = tid; i < 16 * Hc; i += 256) sRh[i] = (_Float16)0.0f;
  __syncthreads();

  for (int t = 0; t < Lc; ++t) {
    // tmp = h_t @ Wh_a + r @ Wr_a ; tt = tanh(r @ Wt_a)
    v8f accM[2] = {}, accT[2] = {};
#pragma unroll 2
    for (int ks = 0; ks < Hc / 32; ++ks) {
      const int k = ks * 32;
      v16h ah = frag_a(Y2 + (size_t)b0 * Lc * Hc + (size_t)t * Hc + k, Lc * Hc, lane);
      v16h ar = frag_a(sRh + k, Hc, lane);
      v16h bf[6];
#pragma unroll
      for (int j2 = 0; j2 < 2; ++j2) {
        const size_t co = (size_t)(2 * wave + j2) * 16 * Hc + k;
        bf[3 * j2 + 0] = frag_bT(WhaT + co, Hc, lane);
        bf[3 * j2 + 1] = frag_bT(WraT + co, Hc, lane);
        bf[3 * j2 + 2] = frag_bT(WtaT + co, Hc, lane);
      }
#pragma unroll
      for (int j2 = 0; j2 < 2; ++j2) {
        accM[j2] = wmma_f16(ah, bf[3 * j2 + 0], accM[j2]);
        accM[j2] = wmma_f16(ar, bf[3 * j2 + 1], accM[j2]);
        accT[j2] = wmma_f16(ar, bf[3 * j2 + 2], accT[j2]);
      }
    }
#pragma unroll
    for (int j2 = 0; j2 < 2; ++j2) {
      const int u = (2 * wave + j2) * 16 + nn;
#pragma unroll
      for (int v = 0; v < 8; ++v) {
        const int m = v + 8 * half;
        sTmp[m * Hc + u] = accM[j2][v];
        sTT[m * Hc + u]  = tanhf(accT[j2][v]);
      }
    }
    __syncthreads();

    // scores: s[row][l] = sum_h tanh(WyY + tmp) * wa + mask
    for (int pidx = tid; pidx < 16 * Lc; pidx += 256) {
      const int row = pidx / Lc, l = pidx % Lc;
      const float* wy = WyY + (size_t)(b0 + row) * Lc * Hc + (size_t)l * Hc;
      float s = 0.0f;
      for (int h = 0; h < Hc; ++h)
        s += tanhf(wy[h] + sTmp[row * Hc + h]) * wa[h];
      if (l >= s_mask[b0 + row]) s -= 10000.0f;
      sScore[row * 64 + l] = s;
    }
    __syncthreads();

    // softmax over l per row
    if (tid < 16) {
      const int row = tid;
      float mx = -1e30f;
      for (int l = 0; l < Lc; ++l) mx = fmaxf(mx, sScore[row * 64 + l]);
      float sum = 0.0f;
      for (int l = 0; l < Lc; ++l) {
        const float e = __expf(sScore[row * 64 + l] - mx);
        sScore[row * 64 + l] = e;
        sum += e;
      }
      const float inv = 1.0f / sum;
      for (int l = 0; l < Lc; ++l) sScore[row * 64 + l] *= inv;
    }
    __syncthreads();

    // r_new = alpha . Y + tt ; record at selection step
    for (int i = tid; i < 16 * Hc; i += 256) {
      const int row = i >> 8, u = i & 255;
      const _Float16* yp = Y + (size_t)(b0 + row) * Lc * Hc + u;
      float acc = 0.0f;
      for (int l = 0; l < Lc; ++l)
        acc += sScore[row * 64 + l] * (float)yp[(size_t)l * Hc];
      const float rn = acc + sTT[i];
      sRh[i] = (_Float16)rn;
      if (t == s_sel[b0 + row] - 1)
        rL[(size_t)(b0 + row) * Hc + u] = (_Float16)rn;
    }
    __syncthreads();
  }
}

// ---- fused final: C (+)= tanh(A1 x B1T^T + A2 x B2T^T), 512x256x256 -------
__global__ __launch_bounds__(32) void gemm2_tanh_kernel(
    const _Float16* __restrict__ A1, const _Float16* __restrict__ B1T,
    const _Float16* __restrict__ A2, const _Float16* __restrict__ B2T,
    float* __restrict__ C, int accumulate) {
  const int lane = threadIdx.x & 31;
  const int nt = blockIdx.x, mt = blockIdx.y;
  v8f accA = {}, accB = {};  // two independent chains
  for (int k = 0; k < Hc; k += 32) {
    v16h a1 = frag_a(A1 + (size_t)mt * 16 * Hc + k, Hc, lane);
    v16h b1 = frag_bT(B1T + (size_t)nt * 16 * Hc + k, Hc, lane);
    v16h a2 = frag_a(A2 + (size_t)mt * 16 * Hc + k, Hc, lane);
    v16h b2 = frag_bT(B2T + (size_t)nt * 16 * Hc + k, Hc, lane);
    accA = wmma_f16(a1, b1, accA);
    accB = wmma_f16(a2, b2, accB);
  }
  const int half = (lane >> 4) & 1, nn = lane & 15;
#pragma unroll
  for (int v = 0; v < 8; ++v) {
    const size_t idx = (size_t)(mt * 16 + v + 8 * half) * Hc + nt * 16 + nn;
    const float val = tanhf(accA[v] + accB[v]);
    C[idx] = accumulate ? (C[idx] + val) : val;
  }
}

__global__ void final_out_kernel(const float* __restrict__ Z,
                                 const float* __restrict__ U,
                                 const float* __restrict__ b_out,
                                 float* __restrict__ out) {
  int idx = blockIdx.x * blockDim.x + threadIdx.x;
  if (idx >= Bc * 2) return;
  const int b = idx >> 1, j = idx & 1;
  float s = b_out[j];
  for (int h = 0; h < Hc; ++h) s += Z[(size_t)b * Hc + h] * U[h * 2 + j];
  out[idx] = s;
}

// ---- orchestration --------------------------------------------------------

extern "C" void kernel_launch(void* const* d_in, const int* in_sizes, int n_in,
                              void* d_out, int out_size, void* d_ws, size_t ws_size,
                              hipStream_t stream) {
  (void)in_sizes; (void)n_in; (void)out_size; (void)ws_size;

  const float* E     = (const float*)d_in[0];
  const float* Wx1   = (const float*)d_in[1];
  const float* Wh1   = (const float*)d_in[2];
  const float* b1    = (const float*)d_in[3];
  const float* Wx2   = (const float*)d_in[4];
  const float* Wh2   = (const float*)d_in[5];
  const float* b2    = (const float*)d_in[6];
  const float* W_y   = (const float*)d_in[7];
  const float* Wh_a  = (const float*)d_in[8];
  const float* Wr_a  = (const float*)d_in[9];
  const float* w_a   = (const float*)d_in[10];
  const float* Wt_a  = (const float*)d_in[11];
  const float* Wp_a  = (const float*)d_in[12];
  const float* Wxa   = (const float*)d_in[13];
  const float* U     = (const float*)d_in[14];
  const float* b_out = (const float*)d_in[15];
  const int*   in1   = (const int*)d_in[16];
  const int*   in2   = (const int*)d_in[17];
  const int*   s1    = (const int*)d_in[18];
  const int*   s2    = (const int*)d_in[19];
  float* out = (float*)d_out;

  char* p = (char*)d_ws;
  auto alloc = [&](size_t bytes) -> void* {
    void* r = (void*)p;
    p += (bytes + 255) & ~(size_t)255;
    return r;
  };

  _Float16* W1T  = (_Float16*)alloc(sizeof(_Float16) * Gc * KT);  // lstm1 fused
  _Float16* W2T  = (_Float16*)alloc(sizeof(_Float16) * Gc * KT);  // lstm2 fused
  _Float16* WyT  = (_Float16*)alloc(sizeof(_Float16) * Hc * Hc);
  _Float16* WhaT = (_Float16*)alloc(sizeof(_Float16) * Hc * Hc);
  _Float16* WraT = (_Float16*)alloc(sizeof(_Float16) * Hc * Hc);
  _Float16* WtaT = (_Float16*)alloc(sizeof(_Float16) * Hc * Hc);
  _Float16* WpaT = (_Float16*)alloc(sizeof(_Float16) * Hc * Hc);
  _Float16* WxaT = (_Float16*)alloc(sizeof(_Float16) * Hc * Hc);
  _Float16* X1h  = (_Float16*)alloc(sizeof(_Float16) * (size_t)Bc * Lc * DPc);
  _Float16* X2h  = (_Float16*)alloc(sizeof(_Float16) * (size_t)Bc * Lc * DPc);
  _Float16* Ya   = (_Float16*)alloc(sizeof(_Float16) * (size_t)Bc * Lc * Hc);
  _Float16* Y2a  = (_Float16*)alloc(sizeof(_Float16) * (size_t)Bc * Lc * Hc);
  _Float16* Yb   = (_Float16*)alloc(sizeof(_Float16) * (size_t)Bc * Lc * Hc);
  _Float16* Y2b  = (_Float16*)alloc(sizeof(_Float16) * (size_t)Bc * Lc * Hc);
  _Float16* hl0  = (_Float16*)alloc(sizeof(_Float16) * (size_t)Bc * Hc);
  _Float16* h2a  = (_Float16*)alloc(sizeof(_Float16) * (size_t)Bc * Hc);
  _Float16* hl2  = (_Float16*)alloc(sizeof(_Float16) * (size_t)Bc * Hc);
  _Float16* h2b  = (_Float16*)alloc(sizeof(_Float16) * (size_t)Bc * Hc);
  float*    WyYa = (float*)alloc(sizeof(float) * (size_t)Bc * Lc * Hc);
  float*    WyYb = (float*)alloc(sizeof(float) * (size_t)Bc * Lc * Hc);
  _Float16* rLa  = (_Float16*)alloc(sizeof(_Float16) * (size_t)Bc * Hc);
  _Float16* rLb  = (_Float16*)alloc(sizeof(_Float16) * (size_t)Bc * Hc);
  float*    Zsum = (float*)alloc(sizeof(float) * (size_t)Bc * Hc);

  const int thr = 256;
  // weight prep (transpose + f16 convert; LSTM weights fused [Wx;pad;Wh])
  build_wT_kernel<<<(Gc * KT + thr - 1) / thr, thr, 0, stream>>>(Wx1, Wh1, W1T);
  build_wT_kernel<<<(Gc * KT + thr - 1) / thr, thr, 0, stream>>>(Wx2, Wh2, W2T);
  cvtT_kernel<<<(Hc * Hc + thr - 1) / thr, thr, 0, stream>>>(W_y,  WyT);
  cvtT_kernel<<<(Hc * Hc + thr - 1) / thr, thr, 0, stream>>>(Wh_a, WhaT);
  cvtT_kernel<<<(Hc * Hc + thr - 1) / thr, thr, 0, stream>>>(Wr_a, WraT);
  cvtT_kernel<<<(Hc * Hc + thr - 1) / thr, thr, 0, stream>>>(Wt_a, WtaT);
  cvtT_kernel<<<(Hc * Hc + thr - 1) / thr, thr, 0, stream>>>(Wp_a, WpaT);
  cvtT_kernel<<<(Hc * Hc + thr - 1) / thr, thr, 0, stream>>>(Wxa,  WxaT);

  // embedding gather (fp32 -> f16, K padded to 320)
  const int nEmb = Bc * Lc * DPc;
  embed_kernel<<<(nEmb + thr - 1) / thr, thr, 0, stream>>>(E, in1, X1h);
  embed_kernel<<<(nEmb + thr - 1) / thr, thr, 0, stream>>>(E, in2, X2h);

  // 4 LSTM runs (branch a: lstm1(x1), lstm2(x2); branch b: lstm1(x2), lstm2(x1))
  lstm_kernel<<<Bc / 16, 256, 0, stream>>>(X1h, W1T, b1, s1, Ya,  hl0);
  lstm_kernel<<<Bc / 16, 256, 0, stream>>>(X2h, W2T, b2, s2, Y2a, h2a);
  lstm_kernel<<<Bc / 16, 256, 0, stream>>>(X2h, W1T, b1, s2, Yb,  hl2);
  lstm_kernel<<<Bc / 16, 256, 0, stream>>>(X1h, W2T, b2, s1, Y2b, h2b);

  // WyY = Y @ W_y  (M = B*L = 30720, N = K = 256)
  dim3 gWy(Hc / 16, (Bc * Lc) / 16);
  gemm_f16f32_kernel<<<gWy, 32, 0, stream>>>(Ya, WyT, WyYa, Hc, Hc, Hc, Hc);
  gemm_f16f32_kernel<<<gWy, 32, 0, stream>>>(Yb, WyT, WyYb, Hc, Hc, Hc, Hc);

  // attention scans
  attend_kernel<<<Bc / 16, 256, 0, stream>>>(Ya, Y2a, WyYa, WhaT, WraT, WtaT,
                                             w_a, s1, s2, rLa);
  attend_kernel<<<Bc / 16, 256, 0, stream>>>(Yb, Y2b, WyYb, WhaT, WraT, WtaT,
                                             w_a, s2, s1, rLb);

  // Zsum = tanh(rLa@Wp_a + h2a@Wxa) + tanh(rLb@Wp_a + h2b@Wxa)
  dim3 gF(Hc / 16, Bc / 16);
  gemm2_tanh_kernel<<<gF, 32, 0, stream>>>(rLa, WpaT, h2a, WxaT, Zsum, 0);
  gemm2_tanh_kernel<<<gF, 32, 0, stream>>>(rLb, WpaT, h2b, WxaT, Zsum, 1);

  // out = Zsum @ U + b_out
  final_out_kernel<<<(Bc * 2 + thr - 1) / thr, thr, 0, stream>>>(Zsum, U, b_out, out);
}
